// DecoderLayer_4277787427226
// MI455X (gfx1250) — matrix operand order repaired and checked by reading, not compiled, not run on hardware
//
#include <hip/hip_runtime.h>
#include <hip/hip_bf16.h>

// ---------------------------------------------------------------------------
// MI455X (gfx1250, wave32) transformer decoder layer.
// Heavy math on v_wmma_f32_16x16x32_f16; GEMM tiles staged with the CDNA5
// async global->LDS DMA path (global_load_async_to_lds_b128 / ASYNCcnt).
// B=4, S=1024, D=1024, H=16, hd=64, FF=4096. M = B*S = 4096 rows.
// ---------------------------------------------------------------------------

typedef _Float16 v16h __attribute__((ext_vector_type(16)));
typedef _Float16 v8h  __attribute__((ext_vector_type(8)));
typedef float    v8f  __attribute__((ext_vector_type(8)));

static __device__ __forceinline__ v8f wmma_f16(v16h a, v16h b, v8f c) {
    return __builtin_amdgcn_wmma_f32_16x16x32_f16(false, a, false, b, (short)0, c, false, false);
}
static __device__ __forceinline__ v16h cat8(v8h lo, v8h hi) {
    return __builtin_shufflevector(lo, hi, 0,1,2,3,4,5,6,7,8,9,10,11,12,13,14,15);
}

// Async 16B global->LDS copy per lane (CDNA5 TDM-adjacent path, ASYNCcnt).
// lds_off: LDS byte offset (low 32 bits of the generic shared pointer —
// the shared aperture keeps the LDS offset in those bits).
static __device__ __forceinline__ void async_g2l_b128(unsigned lds_off, const void* gaddr) {
    asm volatile("global_load_async_to_lds_b128 %0, %1, off"
                 :: "v"(lds_off), "v"(gaddr) : "memory");
}
static __device__ __forceinline__ void wait_async0() {
    asm volatile("s_wait_asynccnt 0x0" ::: "memory");
}

// ---------------------------------------------------------------------------
// fp32 -> f16 convert
// ---------------------------------------------------------------------------
__global__ void k_cvt(const float* __restrict__ in, _Float16* __restrict__ out, int n) {
    for (int i = blockIdx.x * blockDim.x + threadIdx.x; i < n; i += gridDim.x * blockDim.x)
        out[i] = (_Float16)in[i];
}

// ---------------------------------------------------------------------------
// Transpose-convert: W[K][N] fp32 (row major) -> Wt[N][K] f16
// ---------------------------------------------------------------------------
__global__ void k_tr(const float* __restrict__ W, _Float16* __restrict__ Wt, int K, int N) {
    __shared__ float tile[32][33];
    const int n0 = blockIdx.x * 32, k0 = blockIdx.y * 32;
    const int tx = threadIdx.x, ty = threadIdx.y;  // 32 x 8
    #pragma unroll
    for (int i = 0; i < 32; i += 8)
        tile[ty + i][tx] = W[(size_t)(k0 + ty + i) * N + (n0 + tx)];
    __syncthreads();
    #pragma unroll
    for (int i = 0; i < 32; i += 8)
        Wt[(size_t)(n0 + ty + i) * K + (k0 + tx)] = (_Float16)tile[tx][ty + i];
}

// ---------------------------------------------------------------------------
// WMMA GEMM: C[M][N] = A[M][K](f16) * Wt[N][K](f16, pre-transposed) + bias
// Block tile 128x128, BK=64, double-buffered LDS filled by async DMA.
// 256 threads = 8 waves in 4(M) x 2(N); wave tile 32x64 -> 16 wmma / K-tile.
// One s_wait_asynccnt + one barrier per 64-wide K-step.
// ---------------------------------------------------------------------------
#define BK  64
#define LDP 72  // padded LDS row stride (f16), 144B: keeps 16B-aligned v8h slots

__global__ __launch_bounds__(256) void k_gemm(
    const _Float16* __restrict__ A, const _Float16* __restrict__ Bt,
    const float* __restrict__ bias, float* __restrict__ Cf,
    _Float16* __restrict__ Ch, int M, int N, int K, int relu)
{
    __shared__ _Float16 As[2][128][LDP];
    __shared__ _Float16 Bs[2][128][LDP];

    const int tid  = threadIdx.x;
    const int lane = tid & 31, wave = tid >> 5;
    const int wm = wave >> 1, wn = wave & 1;  // 4 x 2 waves
    const int lhalf = lane >> 4, lmod = lane & 15;
    const int bm = blockIdx.y * 128, bn = blockIdx.x * 128;

    v8f acc[2][4];
    #pragma unroll
    for (int i = 0; i < 2; ++i)
        #pragma unroll
        for (int j = 0; j < 4; ++j)
            #pragma unroll
            for (int r = 0; r < 8; ++r) acc[i][j][r] = 0.0f;

    // Stage one 128x64 f16 tile of A and of Bt into LDS buffer `buf` via
    // async DMA: 4 shots x 256 lanes x 16B per matrix.
    auto stage = [&](int kt, int buf) {
        const int k0 = kt * BK;
        #pragma unroll
        for (int i = 0; i < 4; ++i) {
            const int idx = i * 256 + tid;
            const int row = idx >> 3, col = (idx & 7) * 8;
            async_g2l_b128((unsigned)(size_t)&As[buf][row][col],
                           A + (size_t)(bm + row) * K + k0 + col);
            async_g2l_b128((unsigned)(size_t)&Bs[buf][row][col],
                           Bt + (size_t)(bn + row) * K + k0 + col);
        }
    };

    const int nkt = K / BK;
    stage(0, 0);
    for (int kt = 0; kt < nkt; ++kt) {
        const int cur = kt & 1;
        wait_async0();      // my async fills of buffer `cur` have landed
        __syncthreads();    // everyone's have; prior reads of `cur^1` done
        if (kt + 1 < nkt) stage(kt + 1, cur ^ 1);

        #pragma unroll
        for (int kc = 0; kc < 2; ++kc) {
            v16h af[2], bf[4];
            #pragma unroll
            for (int mt = 0; mt < 2; ++mt) {
                const int r = wm * 32 + mt * 16 + lmod;
                // A frag: lanes 0-15 K 0..7 & 16..23; lanes 16-31 K 8..15 & 24..31
                v8h lo = *(const v8h*)&As[cur][r][kc * 32 + lhalf * 8];
                v8h hi = *(const v8h*)&As[cur][r][kc * 32 + 16 + lhalf * 8];
                af[mt] = cat8(lo, hi);
            }
            #pragma unroll
            for (int nt = 0; nt < 4; ++nt) {
                const int r = wn * 64 + nt * 16 + lmod;
                // B frag: lanes 0-15 K 0..7 & 16..23; lanes 16-31 K 8..15 & 24..31, col N = lane%16
                bf[nt] = cat8(*(const v8h*)&Bs[cur][r][kc * 32 + lhalf * 8],
                              *(const v8h*)&Bs[cur][r][kc * 32 + 16 + lhalf * 8]);
            }
            #pragma unroll
            for (int mt = 0; mt < 2; ++mt)
                #pragma unroll
                for (int nt = 0; nt < 4; ++nt)
                    acc[mt][nt] = wmma_f16(af[mt], bf[nt], acc[mt][nt]);
        }
    }

    #pragma unroll
    for (int mt = 0; mt < 2; ++mt)
        #pragma unroll
        for (int nt = 0; nt < 4; ++nt)
            #pragma unroll
            for (int r = 0; r < 8; ++r) {
                const int row = bm + wm * 32 + mt * 16 + r + 8 * lhalf;
                const int col = bn + wn * 64 + nt * 16 + lmod;
                float v = acc[mt][nt][r] + bias[col];
                if (relu) v = fmaxf(v, 0.0f);
                if (Cf) Cf[(size_t)row * N + col] = v;
                if (Ch) Ch[(size_t)row * N + col] = (_Float16)v;
            }
}

// ---------------------------------------------------------------------------
// Flash attention, hd=64, S=1024. Grid (S/128, H, B); 256 threads = 8 waves,
// each wave owns 16 query rows. Streams 32-key blocks:
//   S = Q K^T  (2 wmma per 16-key subtile), online softmax with half-wave
//   shuffle reductions, P re-laid out via per-wave LDS, O += P V (4 wmma).
// q[b,h,s,d] = Q[(b*1024+s)*rs + h*hs + d]; output merged heads f16.
// ---------------------------------------------------------------------------
__global__ __launch_bounds__(256) void k_attn(
    const _Float16* __restrict__ Q,  int q_rs, int q_hs,
    const _Float16* __restrict__ Kp, int k_rs, int k_hs,
    const _Float16* __restrict__ Vp, int v_rs, int v_hs,
    _Float16* __restrict__ O, int causal)
{
    __shared__ _Float16 Ks[32][64 + 16];   // [keypos][hd]
    __shared__ _Float16 Vts[64][32 + 16];  // [hd][keypos]  (V transposed)
    __shared__ _Float16 Pw[8][16][32];     // per-wave P tile round-trip

    const int tid = threadIdx.x, lane = tid & 31, wave = tid >> 5;
    const int lhalf = lane >> 4, lmod = lane & 15;
    const int qblk = blockIdx.x, h = blockIdx.y, b = blockIdx.z;
    const int qrow = qblk * 128 + wave * 16;
    const size_t brow = (size_t)b * 1024;

    v16h qf[2];
    {
        const _Float16* qp = Q + (brow + qrow + lmod) * q_rs + h * q_hs;
        #pragma unroll
        for (int kc = 0; kc < 2; ++kc) {
            const _Float16* p = qp + kc * 32 + lhalf * 8;
            qf[kc] = cat8(*(const v8h*)p, *(const v8h*)(p + 16));
        }
    }

    v8f o[4];
    #pragma unroll
    for (int j = 0; j < 4; ++j)
        #pragma unroll
        for (int r = 0; r < 8; ++r) o[j][r] = 0.0f;
    float m8[8], l8[8];
    #pragma unroll
    for (int r = 0; r < 8; ++r) { m8[r] = -1e30f; l8[r] = 0.0f; }

    const int nkb = causal ? (qblk * 4 + 4) : 32;
    for (int kb = 0; kb < nkb; ++kb) {
        __syncthreads();
        {   // stage K tile (row major) and V tile (transposed) into LDS
            const int krow = tid >> 3, c = (tid & 7) * 8;
            const _Float16* kp = Kp + (brow + kb * 32 + krow) * k_rs + h * k_hs + c;
            *(v8h*)&Ks[krow][c] = *(const v8h*)kp;
            const _Float16* vp = Vp + (brow + kb * 32 + krow) * v_rs + h * v_hs + c;
            v8h vv = *(const v8h*)vp;
            #pragma unroll
            for (int i = 0; i < 8; ++i) Vts[c + i][krow] = vv[i];
        }
        __syncthreads();

        v8f s[2];
        #pragma unroll
        for (int nt = 0; nt < 2; ++nt) {
            #pragma unroll
            for (int r = 0; r < 8; ++r) s[nt][r] = 0.0f;
            #pragma unroll
            for (int kc = 0; kc < 2; ++kc) {
                v16h bk = cat8(*(const v8h*)&Ks[nt * 16 + lmod][kc * 32 + lhalf * 8],
                               *(const v8h*)&Ks[nt * 16 + lmod][kc * 32 + 16 + lhalf * 8]);
                s[nt] = wmma_f16(qf[kc], bk, s[nt]);
            }
        }

        #pragma unroll
        for (int r = 0; r < 8; ++r) {
            const int row = qrow + r + 8 * lhalf;
            float s0 = s[0][r] * 0.125f;   // 1/sqrt(64)
            float s1 = s[1][r] * 0.125f;
            if (causal) {
                const int c0 = kb * 32 + lmod;
                if (c0 > row)      s0 = -1e30f;
                if (c0 + 16 > row) s1 = -1e30f;
            }
            float mx = fmaxf(s0, s1);
            #pragma unroll
            for (int off = 8; off; off >>= 1) mx = fmaxf(mx, __shfl_xor(mx, off, 32));
            const float mnew  = fmaxf(m8[r], mx);
            const float alpha = __expf(m8[r] - mnew);
            const float p0 = __expf(s0 - mnew);
            const float p1 = __expf(s1 - mnew);
            float sum = p0 + p1;
            #pragma unroll
            for (int off = 8; off; off >>= 1) sum += __shfl_xor(sum, off, 32);
            l8[r] = l8[r] * alpha + sum;
            m8[r] = mnew;
            Pw[wave][r + 8 * lhalf][lmod]      = (_Float16)p0;
            Pw[wave][r + 8 * lhalf][16 + lmod] = (_Float16)p1;
            #pragma unroll
            for (int j = 0; j < 4; ++j) o[j][r] *= alpha;
        }

        // re-read P as an A fragment (same-wave LDS ops are in order)
        v8h plo = *(const v8h*)&Pw[wave][lmod][lhalf * 8];
        v8h phi = *(const v8h*)&Pw[wave][lmod][16 + lhalf * 8];
        v16h pa = cat8(plo, phi);
        #pragma unroll
        for (int j = 0; j < 4; ++j) {
            v16h bv = cat8(*(const v8h*)&Vts[j * 16 + lmod][lhalf * 8],
                           *(const v8h*)&Vts[j * 16 + lmod][16 + lhalf * 8]);
            o[j] = wmma_f16(pa, bv, o[j]);
        }
    }

    #pragma unroll
    for (int j = 0; j < 4; ++j)
        #pragma unroll
        for (int r = 0; r < 8; ++r) {
            const int row = qrow + r + 8 * lhalf;
            const int col = h * 64 + j * 16 + lmod;
            O[(brow + row) * 1024 + col] = (_Float16)(o[j][r] / l8[r]);
        }
}

// ---------------------------------------------------------------------------
// out = LayerNorm(A + R) * g + be ; writes f32 and optional f16 copy.
// ---------------------------------------------------------------------------
__global__ __launch_bounds__(256) void k_ln(
    const float* __restrict__ A, const float* __restrict__ R,
    const float* __restrict__ g, const float* __restrict__ be,
    float* __restrict__ out32, _Float16* __restrict__ out16)
{
    __shared__ float red[256];
    const int row = blockIdx.x, tid = threadIdx.x;
    const size_t base = (size_t)row * 1024;
    float x[4], s = 0.0f;
    #pragma unroll
    for (int i = 0; i < 4; ++i) {
        x[i] = A[base + tid + i * 256] + R[base + tid + i * 256];
        s += x[i];
    }
    red[tid] = s;
    __syncthreads();
    for (int off = 128; off; off >>= 1) {
        if (tid < off) red[tid] += red[tid + off];
        __syncthreads();
    }
    const float mean = red[0] * (1.0f / 1024.0f);
    __syncthreads();
    float s2 = 0.0f;
    #pragma unroll
    for (int i = 0; i < 4; ++i) { const float d = x[i] - mean; s2 += d * d; }
    red[tid] = s2;
    __syncthreads();
    for (int off = 128; off; off >>= 1) {
        if (tid < off) red[tid] += red[tid + off];
        __syncthreads();
    }
    const float rstd = rsqrtf(red[0] * (1.0f / 1024.0f) + 1e-5f);
    #pragma unroll
    for (int i = 0; i < 4; ++i) {
        const int c = tid + i * 256;
        const float v = (x[i] - mean) * rstd * g[c] + be[c];
        out32[base + c] = v;
        if (out16) out16[base + c] = (_Float16)v;
    }
}

// ---------------------------------------------------------------------------
// Host orchestration
// ---------------------------------------------------------------------------
extern "C" void kernel_launch(void* const* d_in, const int* in_sizes, int n_in,
                              void* d_out, int out_size, void* d_ws, size_t ws_size,
                              hipStream_t stream)
{
    const int M = 4096, D = 1024, FF = 4096;
    const float* x    = (const float*)d_in[0];
    const float* y    = (const float*)d_in[1];
    const float* Wqkv = (const float*)d_in[4];
    const float* bqkv = (const float*)d_in[5];
    const float* Wo1  = (const float*)d_in[6];
    const float* bo1  = (const float*)d_in[7];
    const float* Wkv  = (const float*)d_in[8];
    const float* bkv  = (const float*)d_in[9];
    const float* Wq   = (const float*)d_in[10];
    const float* bq   = (const float*)d_in[11];
    const float* Wo2  = (const float*)d_in[12];
    const float* bo2  = (const float*)d_in[13];
    const float* W1   = (const float*)d_in[14];
    const float* b1   = (const float*)d_in[15];
    const float* W2   = (const float*)d_in[16];
    const float* b2   = (const float*)d_in[17];
    const float* g1   = (const float*)d_in[18];
    const float* be1  = (const float*)d_in[19];
    const float* g2   = (const float*)d_in[20];
    const float* be2  = (const float*)d_in[21];
    const float* g3   = (const float*)d_in[22];
    const float* be3  = (const float*)d_in[23];

    char* ws = (char*)d_ws;
    size_t off = 0;
    auto alloc = [&](size_t bytes) -> void* {
        void* p = ws + off;
        off += (bytes + 255) & ~(size_t)255;
        return p;
    };

    _Float16* yh     = (_Float16*)alloc((size_t)M * D * 2);
    _Float16* xh     = (_Float16*)alloc((size_t)M * D * 2);
    _Float16* Wqkv_t = (_Float16*)alloc((size_t)3 * D * D * 2);
    _Float16* Wo1_t  = (_Float16*)alloc((size_t)D * D * 2);
    _Float16* Wkv_t  = (_Float16*)alloc((size_t)2 * D * D * 2);
    _Float16* Wq_t   = (_Float16*)alloc((size_t)D * D * 2);
    _Float16* Wo2_t  = (_Float16*)alloc((size_t)D * D * 2);
    _Float16* W1_t   = (_Float16*)alloc((size_t)FF * D * 2);
    _Float16* W2_t   = (_Float16*)alloc((size_t)D * FF * 2);
    _Float16* qkv_h  = (_Float16*)alloc((size_t)M * 3 * D * 2);
    _Float16* kv_h   = (_Float16*)alloc((size_t)M * 2 * D * 2);
    _Float16* q2_h   = (_Float16*)alloc((size_t)M * D * 2);
    _Float16* attn_h = (_Float16*)alloc((size_t)M * D * 2);  // reused: self then cross
    float*    o32    = (float*)   alloc((size_t)M * D * 4);  // reused: o1, o2, ff2
    float*    y1f    = (float*)   alloc((size_t)M * D * 4);
    _Float16* y1h    = (_Float16*)alloc((size_t)M * D * 2);
    float*    y2f    = (float*)   alloc((size_t)M * D * 4);
    _Float16* y2h    = (_Float16*)alloc((size_t)M * D * 2);
    _Float16* ff1_h  = (_Float16*)alloc((size_t)M * FF * 2);
    (void)ws_size; (void)in_sizes; (void)n_in; (void)out_size;

    const dim3 trb(32, 8);
    k_cvt<<<2048, 256, 0, stream>>>(y, yh, M * D);
    k_cvt<<<2048, 256, 0, stream>>>(x, xh, M * D);
    k_tr<<<dim3(3 * D / 32, D / 32), trb, 0, stream>>>(Wqkv, Wqkv_t, D, 3 * D);
    k_tr<<<dim3(D / 32, D / 32),     trb, 0, stream>>>(Wo1,  Wo1_t,  D, D);
    k_tr<<<dim3(2 * D / 32, D / 32), trb, 0, stream>>>(Wkv,  Wkv_t,  D, 2 * D);
    k_tr<<<dim3(D / 32, D / 32),     trb, 0, stream>>>(Wq,   Wq_t,   D, D);
    k_tr<<<dim3(D / 32, D / 32),     trb, 0, stream>>>(Wo2,  Wo2_t,  D, D);
    k_tr<<<dim3(FF / 32, D / 32),    trb, 0, stream>>>(W1,   W1_t,   D, FF);
    k_tr<<<dim3(D / 32, FF / 32),    trb, 0, stream>>>(W2,   W2_t,   FF, D);

    // --- self attention on y ---
    k_gemm<<<dim3(3 * D / 128, M / 128), 256, 0, stream>>>(
        yh, Wqkv_t, bqkv, (float*)nullptr, qkv_h, M, 3 * D, D, 0);
    // qkv column layout: head h occupies cols [h*192, h*192+192): q|k|v of 64
    k_attn<<<dim3(8, 16, 4), 256, 0, stream>>>(
        qkv_h,        3 * D, 192,
        qkv_h + 64,   3 * D, 192,
        qkv_h + 128,  3 * D, 192,
        attn_h, /*causal=*/1);
    k_gemm<<<dim3(D / 128, M / 128), 256, 0, stream>>>(
        attn_h, Wo1_t, bo1, o32, (_Float16*)nullptr, M, D, D, 0);
    k_ln<<<M, 256, 0, stream>>>(o32, y, g1, be1, y1f, y1h);

    // --- cross attention: k,v from x; q from y1 ---
    k_gemm<<<dim3(2 * D / 128, M / 128), 256, 0, stream>>>(
        xh, Wkv_t, bkv, (float*)nullptr, kv_h, M, 2 * D, D, 0);
    k_gemm<<<dim3(D / 128, M / 128), 256, 0, stream>>>(
        y1h, Wq_t, bq, (float*)nullptr, q2_h, M, D, D, 0);
    k_attn<<<dim3(8, 16, 4), 256, 0, stream>>>(
        q2_h,        D,     64,
        kv_h,        2 * D, 128,
        kv_h + 64,   2 * D, 128,
        attn_h, /*causal=*/0);
    k_gemm<<<dim3(D / 128, M / 128), 256, 0, stream>>>(
        attn_h, Wo2_t, bo2, o32, (_Float16*)nullptr, M, D, D, 0);
    k_ln<<<M, 256, 0, stream>>>(o32, y1f, g2, be2, y2f, y2h);

    // --- FFN ---
    k_gemm<<<dim3(FF / 128, M / 128), 256, 0, stream>>>(
        y2h, W1_t, b1, (float*)nullptr, ff1_h, M, FF, D, /*relu=*/1);
    k_gemm<<<dim3(D / 128, M / 128), 256, 0, stream>>>(
        ff1_h, W2_t, b2, o32, (_Float16*)nullptr, M, D, FF, 0);
    k_ln<<<M, 256, 0, stream>>>(o32, y2f, g3, be3, (float*)d_out, (_Float16*)nullptr);
}
